// Decoder_39118562132545
// MI455X (gfx1250) — compile-verified, matrix operand order checked by reading, not run on hardware
//
#include <hip/hip_runtime.h>
#include <hip/hip_bf16.h>

#define B_   128
#define S_   2048
#define E_   1024
#define H_   1024
#define A_   512
#define EMB_ 512
#define V_   32000

typedef __attribute__((ext_vector_type(16))) __bf16 v16bf;
typedef __attribute__((ext_vector_type(8)))  float  v8f;
typedef unsigned int u32x4 __attribute__((ext_vector_type(4)));
typedef int          i32x4 __attribute__((ext_vector_type(4)));
typedef int          i32x8 __attribute__((ext_vector_type(8)));

__device__ __forceinline__ unsigned short f2bf(float f) {
  union { float f; unsigned u; } x; x.f = f;
  unsigned r = x.u + 0x7FFFu + ((x.u >> 16) & 1u);   // round-to-nearest-even
  return (unsigned short)(r >> 16);
}

union FragBF { v16bf v; uint4 q[2]; unsigned short u[16]; };

// ---------------------------------------------------------------------------
// TDM: issue a 2D tensor_load_to_lds. D# built per CDNA5 ISA ch.8:
//  group0: count=1 [1:0], lds_addr [63:32], global_addr [120:64], type=2 [127:126]
//  group1: data_size=4B (code 2) [17:16], tensor_dim0 [79:48], tensor_dim1 [111:80],
//          tile_dim0 [127:112], tile_dim1 [143:128], tensor_dim0_stride [207:160]
// ---------------------------------------------------------------------------
__device__ __forceinline__ void tdm_load_2d_f32(const void* gptr, unsigned lds_off,
                                                unsigned tile_d0, unsigned tile_d1,
                                                unsigned tensor_d0, unsigned tensor_d1,
                                                unsigned stride0)
{
  unsigned long long ga = (unsigned long long)gptr;
  u32x4 g0;
  g0.x = 1u;                                   // count=1 (valid user descriptor)
  g0.y = lds_off;                              // LDS byte address
  g0.z = (unsigned)ga;                         // global addr lo
  g0.w = (unsigned)((ga >> 32) & 0x01FFFFFFu) | (2u << 30);   // addr hi | type=2

  unsigned long long d0 = (2ull << 16)                          // data_size = 4B
                        | ((unsigned long long)tensor_d0 << 48); // tensor_dim0 lo16
  unsigned long long d1 = ((unsigned long long)(tensor_d0 >> 16))       // dim0 hi
                        | ((unsigned long long)tensor_d1 << 16)          // tensor_dim1
                        | ((unsigned long long)tile_d0 << 48);           // tile_dim0
  unsigned long long d2 = (unsigned long long)tile_d1                    // tile_dim1
                        | ((unsigned long long)stride0 << 32);           // dim0 stride
  unsigned long long d3 = 0;
  i32x8 g1 = { (int)d0, (int)(d0 >> 32), (int)d1, (int)(d1 >> 32),
               (int)d2, (int)(d2 >> 32), (int)d3, (int)(d3 >> 32) };
  i32x4 z4 = { 0, 0, 0, 0 };
#if defined(__clang_major__) && (__clang_major__ >= 23)
  i32x8 z8 = { 0, 0, 0, 0, 0, 0, 0, 0 };
  __builtin_amdgcn_tensor_load_to_lds(g0, g1, z4, z4, z8, 0);
#else
  __builtin_amdgcn_tensor_load_to_lds(g0, g1, z4, z4, 0);
#endif
}

// ---------------------------------------------------------------------------
// Generic C[128,N] = A[128,K] @ W[N,K]^T (+bias), f32 in/out, bf16 WMMA core.
// Block: 256 threads = 8 waves (4 M-waves x 2 N-waves); tile 128M x 128N x 32K.
// ---------------------------------------------------------------------------
__global__ __launch_bounds__(256)
void gemm_bf16_kernel(const float* __restrict__ A, int lda,
                      const float* __restrict__ W, int ldw,
                      const float* __restrict__ bias,
                      float* __restrict__ C, int ldc, int K)
{
  __shared__ unsigned short As[128][40];   // 32 K + 8 pad
  __shared__ unsigned short Bs[128][40];
  const int tid  = threadIdx.x;
  const int lane = tid & 31;
  const int wid  = tid >> 5;
  const int wm   = wid & 3;                 // 32 rows each
  const int wn   = wid >> 2;                // 64 cols each
  const int n0   = blockIdx.x * 128;
  const int lr    = lane & 15;
  const int lhalf = (lane < 16) ? 0 : 8;    // ISA 16-bit A/B operand K-split

  v8f zero = {};
  v8f acc[2][4];
  for (int mi = 0; mi < 2; ++mi)
    for (int ni = 0; ni < 4; ++ni) acc[mi][ni] = zero;

  for (int k0 = 0; k0 < K; k0 += 32) {
    // stage f32 -> bf16 LDS tiles (128 x 32 each, float4 vectorized)
    for (int p = 0; p < 4; ++p) {
      int idx = tid + p * 256;              // 1024 float4 per matrix
      int row = idx >> 3;
      int c4  = (idx & 7) * 4;
      float4 av = *(const float4*)(A + (size_t)row * lda + k0 + c4);
      As[row][c4+0] = f2bf(av.x); As[row][c4+1] = f2bf(av.y);
      As[row][c4+2] = f2bf(av.z); As[row][c4+3] = f2bf(av.w);
      float4 bv = *(const float4*)(W + (size_t)(n0 + row) * ldw + k0 + c4);
      Bs[row][c4+0] = f2bf(bv.x); Bs[row][c4+1] = f2bf(bv.y);
      Bs[row][c4+2] = f2bf(bv.z); Bs[row][c4+3] = f2bf(bv.w);
    }
    // prefetch next K tile (global_prefetch_b8) to keep the weight stream ahead
    if (k0 + 32 < K) {
      int row = tid >> 1, half = (tid & 1) * 16;
      __builtin_prefetch(A + (size_t)row * lda + k0 + 32 + half, 0, 0);
      __builtin_prefetch(W + (size_t)(n0 + row) * ldw + k0 + 32 + half, 0, 0);
    }
    __syncthreads();
    FragBF af[2], bf[4];
    for (int mi = 0; mi < 2; ++mi) {
      int r = wm * 32 + mi * 16 + lr;
      af[mi].q[0] = *(const uint4*)&As[r][lhalf];
      af[mi].q[1] = *(const uint4*)&As[r][lhalf + 16];
    }
    for (int ni = 0; ni < 4; ++ni) {
      int r = wn * 64 + ni * 16 + lr;
      bf[ni].q[0] = *(const uint4*)&Bs[r][lhalf];
      bf[ni].q[1] = *(const uint4*)&Bs[r][lhalf + 16];
    }
    for (int mi = 0; mi < 2; ++mi)
      for (int ni = 0; ni < 4; ++ni)
        acc[mi][ni] = __builtin_amdgcn_wmma_f32_16x16x32_bf16(
            false, af[mi].v, false, bf[ni].v, (short)0, acc[mi][ni], false, false);
    __syncthreads();
  }

  const int rbase = (lane >= 16) ? 8 : 0;   // C/D VGPR layout (ISA 7.12.2)
  for (int mi = 0; mi < 2; ++mi)
    for (int ni = 0; ni < 4; ++ni)
      for (int i = 0; i < 8; ++i) {
        int row = wm * 32 + mi * 16 + rbase + i;
        int col = n0 + wn * 64 + ni * 16 + lr;
        float v = acc[mi][ni][i];
        if (bias) v += bias[col];
        C[(size_t)row * ldc + col] = v;
      }
}

// ---------------------------------------------------------------------------
// Fused attention scores: scores[b,s] = sum_a Ws[a]*tanh(q[b,a] + enc[b,s,:]@Wk[a,:])
// Workgroup: one b, 64 s-rows. enc tile streamed by the Tensor Data Mover into a
// double-buffered f32 staging region, compacted LDS-locally to a 132KB bf16 tile
// (total ~260KB LDS -- CDNA5's 320KB/WGP). Keys live only in VGPRs.
// ---------------------------------------------------------------------------
#define EPAD 1032
#define ENCS_BYTES   (64 * EPAD * 2)            // 132096
#define STG_BYTES    (16 * 1024 * 4)            // 65536 per buffer
__global__ __launch_bounds__(256)
void attn_scores_kernel(const float* __restrict__ enc,   // [B,S,E]
                        const float* __restrict__ Wk,    // [A,E]
                        const float* __restrict__ Ws,    // [A]
                        const float* __restrict__ q,     // [B,A]
                        float* __restrict__ scores)      // [B,S]
{
  extern __shared__ char smem[];
  unsigned short (*encs)[EPAD] = (unsigned short(*)[EPAD])smem;
  float* stg0 = (float*)(smem + ENCS_BYTES);
  float* stg1 = (float*)(smem + ENCS_BYTES + STG_BYTES);
  float* qs   = (float*)(smem + ENCS_BYTES + 2 * STG_BYTES);   // [A_]
  float* sc   = qs + A_;                                       // [64]

  const int tid = threadIdx.x, lane = tid & 31, wid = tid >> 5;
  const int wm = wid & 3, wn = wid >> 2;
  const int b  = blockIdx.y;
  const int s0 = blockIdx.x * 64;
  const int lr = lane & 15, lhalf = (lane < 16) ? 0 : 8;

  const float* ebase = enc + ((size_t)b * S_ + s0) * E_;
  const unsigned stg_off[2] = {
    (unsigned)(size_t)(void*)stg0,
    (unsigned)(size_t)(void*)stg1 };

  for (int i = tid; i < A_; i += 256) qs[i] = q[(size_t)b * A_ + i];
  if (tid < 64) sc[tid] = 0.0f;

  // TDM pipeline: DMA 16-row groups of the enc tile, convert f32->bf16 in LDS.
  if (wid == 0) {
    tdm_load_2d_f32(ebase, stg_off[0], E_, 16, E_, S_ * E_, E_);
    __builtin_amdgcn_s_wait_tensorcnt((short)0);
  }
  __syncthreads();
  for (int g = 0; g < 4; ++g) {
    if (wid == 0 && g + 1 < 4)
      tdm_load_2d_f32(ebase + (size_t)(g + 1) * 16 * E_, stg_off[(g + 1) & 1],
                      E_, 16, E_, S_ * E_, E_);
    const float* sb = (g & 1) ? stg1 : stg0;
    for (int p = 0; p < 16; ++p) {           // 4096 float4 per group
      int idx = tid + p * 256;
      int row = idx >> 8;                    // 0..15
      int c4  = (idx & 255) * 4;
      float4 v = *(const float4*)(sb + (size_t)row * E_ + c4);
      unsigned short* dst = &encs[g * 16 + row][c4];
      dst[0] = f2bf(v.x); dst[1] = f2bf(v.y);
      dst[2] = f2bf(v.z); dst[3] = f2bf(v.w);
    }
    if (wid == 0 && g + 1 < 4)
      __builtin_amdgcn_s_wait_tensorcnt((short)0);
    __syncthreads();
  }

  for (int a0 = 0; a0 < A_; a0 += 128) {
    v8f zero = {};
    v8f acc[4];
    for (int ni = 0; ni < 4; ++ni) acc[ni] = zero;

    for (int e0 = 0; e0 < E_; e0 += 32) {
      FragBF af;
      int r = wm * 16 + lr;
      af.q[0] = *(const uint4*)&encs[r][e0 + lhalf];
      af.q[1] = *(const uint4*)&encs[r][e0 + lhalf + 16];
      for (int ni = 0; ni < 4; ++ni) {
        int n = a0 + wn * 64 + ni * 16 + lr;
        const float* wp = Wk + (size_t)n * E_ + e0 + lhalf;
        float4 w0 = *(const float4*)(wp);
        float4 w1 = *(const float4*)(wp + 4);
        float4 w2 = *(const float4*)(wp + 16);
        float4 w3 = *(const float4*)(wp + 20);
        FragBF bfr;
        bfr.u[0]  = f2bf(w0.x); bfr.u[1]  = f2bf(w0.y);
        bfr.u[2]  = f2bf(w0.z); bfr.u[3]  = f2bf(w0.w);
        bfr.u[4]  = f2bf(w1.x); bfr.u[5]  = f2bf(w1.y);
        bfr.u[6]  = f2bf(w1.z); bfr.u[7]  = f2bf(w1.w);
        bfr.u[8]  = f2bf(w2.x); bfr.u[9]  = f2bf(w2.y);
        bfr.u[10] = f2bf(w2.z); bfr.u[11] = f2bf(w2.w);
        bfr.u[12] = f2bf(w3.x); bfr.u[13] = f2bf(w3.y);
        bfr.u[14] = f2bf(w3.z); bfr.u[15] = f2bf(w3.w);
        acc[ni] = __builtin_amdgcn_wmma_f32_16x16x32_bf16(
            false, af.v, false, bfr.v, (short)0, acc[ni], false, false);
      }
    }
    // epilogue: Ws * tanh(q + key) folded into per-row partial sums
    const int rbase = (lane >= 16) ? 8 : 0;
    for (int ni = 0; ni < 4; ++ni) {
      int a_col = a0 + wn * 64 + ni * 16 + lr;
      float wsv = Ws[a_col];
      float qv  = qs[a_col];
      for (int i = 0; i < 8; ++i) {
        int srow = wm * 16 + rbase + i;
        atomicAdd(&sc[srow], wsv * tanhf(qv + acc[ni][i]));
      }
    }
  }
  __syncthreads();
  if (tid < 64) scores[(size_t)b * S_ + s0 + tid] = sc[tid];
}

// ---------------------------------------------------------------------------
__global__ __launch_bounds__(256)
void softmax_kernel(const float* __restrict__ scores,
                    const unsigned char* __restrict__ mask,
                    float* __restrict__ weights)
{
  __shared__ float red[256];
  int b = blockIdx.x, tid = threadIdx.x;
  const float* srow = scores + (size_t)b * S_;
  const unsigned char* mrow = mask + (size_t)b * S_;
  float m = -1e30f;
  for (int s = tid; s < S_; s += 256) {
    float v = mrow[s] ? srow[s] : -1e9f;
    m = fmaxf(m, v);
  }
  red[tid] = m; __syncthreads();
  for (int off = 128; off > 0; off >>= 1) {
    if (tid < off) red[tid] = fmaxf(red[tid], red[tid + off]);
    __syncthreads();
  }
  m = red[0]; __syncthreads();
  float sum = 0.f;
  for (int s = tid; s < S_; s += 256) {
    float v = mrow[s] ? srow[s] : -1e9f;
    sum += __expf(v - m);
  }
  red[tid] = sum; __syncthreads();
  for (int off = 128; off > 0; off >>= 1) {
    if (tid < off) red[tid] += red[tid + off];
    __syncthreads();
  }
  float inv = 1.0f / red[0];
  for (int s = tid; s < S_; s += 256) {
    float v = mrow[s] ? srow[s] : -1e9f;
    weights[(size_t)b * S_ + s] = __expf(v - m) * inv;
  }
}

// context[b,e] = sum_s weights[b,s] * enc[b,s,e]
__global__ __launch_bounds__(256)
void context_kernel(const float* __restrict__ weights,
                    const float* __restrict__ enc,
                    float* __restrict__ ctx)
{
  __shared__ float wls[256];
  int b = blockIdx.y;
  int e = blockIdx.x * 256 + threadIdx.x;
  float acc = 0.f;
  for (int sc0 = 0; sc0 < S_; sc0 += 256) {
    __syncthreads();
    wls[threadIdx.x] = weights[(size_t)b * S_ + sc0 + threadIdx.x];
    __syncthreads();
    const float* eb = enc + ((size_t)b * S_ + sc0) * E_ + e;
    for (int ss = 0; ss < 256; ++ss) acc += wls[ss] * eb[(size_t)ss * E_];
  }
  ctx[(size_t)b * E_ + e] = acc;
}

__global__ void embed_kernel(const int* __restrict__ tok,
                             const float* __restrict__ table,
                             float* __restrict__ out)
{
  int b = blockIdx.x;
  int t = tok[b];
  for (int e = threadIdx.x; e < EMB_; e += blockDim.x)
    out[(size_t)b * EMB_ + e] = table[(size_t)t * EMB_ + e];
}

__global__ void concat_x_kernel(const float* __restrict__ emb,
                                const float* __restrict__ ctx,
                                float* __restrict__ x)
{
  int idx = blockIdx.x * 256 + threadIdx.x;
  if (idx >= B_ * (EMB_ + E_)) return;
  int b = idx / (EMB_ + E_), j = idx % (EMB_ + E_);
  x[idx] = (j < EMB_) ? emb[(size_t)b * EMB_ + j]
                      : ctx[(size_t)b * E_ + (j - EMB_)];
}

__global__ void concat_out_kernel(const float* __restrict__ h1,
                                  const float* __restrict__ ctx,
                                  const float* __restrict__ emb,
                                  float* __restrict__ xo)
{
  const int W = H_ + E_ + EMB_;            // 2560
  int idx = blockIdx.x * 256 + threadIdx.x;
  if (idx >= B_ * W) return;
  int b = idx / W, j = idx % W;
  float v;
  if (j < H_)            v = h1[(size_t)b * H_ + j];
  else if (j < H_ + E_)  v = ctx[(size_t)b * E_ + (j - H_)];
  else                   v = emb[(size_t)b * EMB_ + (j - H_ - E_)];
  xo[idx] = v;
}

__global__ void gru_gates_kernel(const float* __restrict__ gi,
                                 const float* __restrict__ gh,
                                 const float* __restrict__ hprev,
                                 float* __restrict__ hout)
{
  int idx = blockIdx.x * 256 + threadIdx.x;   // B_*H_
  if (idx >= B_ * H_) return;
  int b = idx / H_, j = idx % H_;
  const float* gib = gi + (size_t)b * 3 * H_;
  const float* ghb = gh + (size_t)b * 3 * H_;
  float ir = gib[j], iz = gib[H_ + j], in = gib[2 * H_ + j];
  float hr = ghb[j], hz = ghb[H_ + j], hn = ghb[2 * H_ + j];
  float r = 1.f / (1.f + __expf(-(ir + hr)));
  float z = 1.f / (1.f + __expf(-(iz + hz)));
  float n = tanhf(in + r * hn);
  hout[idx] = (1.f - z) * n + z * hprev[idx];
}

// ---------------------------------------------------------------------------
extern "C" void kernel_launch(void* const* d_in, const int* in_sizes, int n_in,
                              void* d_out, int out_size, void* d_ws, size_t ws_size,
                              hipStream_t stream)
{
  (void)in_sizes; (void)n_in; (void)out_size; (void)ws_size;
  const int*   tok   = (const int*)d_in[0];
  const float* hs    = (const float*)d_in[1];     // [2,B,H]
  const float* enc   = (const float*)d_in[2];     // [B,S,E]
  const unsigned char* mask = (const unsigned char*)d_in[3];
  const float* table = (const float*)d_in[4];
  const float* Wq    = (const float*)d_in[5];
  const float* Wk    = (const float*)d_in[6];
  const float* Ws    = (const float*)d_in[7];
  const float* g0Wih = (const float*)d_in[8];
  const float* g0Whh = (const float*)d_in[9];
  const float* g0bih = (const float*)d_in[10];
  const float* g0bhh = (const float*)d_in[11];
  const float* g1Wih = (const float*)d_in[12];
  const float* g1Whh = (const float*)d_in[13];
  const float* g1bih = (const float*)d_in[14];
  const float* g1bhh = (const float*)d_in[15];
  const float* Wout  = (const float*)d_in[16];
  const float* bout  = (const float*)d_in[17];

  float* out    = (float*)d_out;
  float* logits = out;                                   // [B,V]
  float* h0_out = out + (size_t)B_ * V_;                 // [B,H]
  float* h1_out = h0_out + (size_t)B_ * H_;              // [B,H]
  float* w_out  = h1_out + (size_t)B_ * H_;              // [B,S]

  float* wsf = (float*)d_ws;
  float* emb = wsf;                  // 128*512
  float* qry = emb + 65536;          // 128*512
  float* scr = qry + 65536;          // 128*2048
  float* ctx = scr + 262144;         // 128*1024
  float* xin = ctx + 131072;         // 128*1536
  float* gi  = xin + 196608;         // 128*3072
  float* gh  = gi  + 393216;         // 128*3072
  float* xo  = gh  + 393216;         // 128*2560

  const float* hs0 = hs;
  const float* hs1 = hs + (size_t)B_ * H_;

  embed_kernel<<<B_, 256, 0, stream>>>(tok, table, emb);

  // query = hs[-1] @ Wq^T  -> [B, A]
  gemm_bf16_kernel<<<A_ / 128, 256, 0, stream>>>(hs1, H_, Wq, H_, nullptr, qry, A_, H_);

  // fused keys/tanh/dot: TDM-staged 132KB enc tile + 128KB staging in LDS
  size_t smem = (size_t)ENCS_BYTES + 2 * STG_BYTES + (size_t)A_ * 4 + 64 * 4;
  attn_scores_kernel<<<dim3(S_ / 64, B_), 256, smem, stream>>>(enc, Wk, Ws, qry, scr);

  softmax_kernel<<<B_, 256, 0, stream>>>(scr, mask, w_out);
  context_kernel<<<dim3(E_ / 256, B_), 256, 0, stream>>>(w_out, enc, ctx);

  concat_x_kernel<<<(B_ * (EMB_ + E_) + 255) / 256, 256, 0, stream>>>(emb, ctx, xin);

  // GRU layer 0
  gemm_bf16_kernel<<<(3 * H_) / 128, 256, 0, stream>>>(xin, EMB_ + E_, g0Wih, EMB_ + E_, g0bih, gi, 3 * H_, EMB_ + E_);
  gemm_bf16_kernel<<<(3 * H_) / 128, 256, 0, stream>>>(hs0, H_, g0Whh, H_, g0bhh, gh, 3 * H_, H_);
  gru_gates_kernel<<<(B_ * H_ + 255) / 256, 256, 0, stream>>>(gi, gh, hs0, h0_out);

  // GRU layer 1
  gemm_bf16_kernel<<<(3 * H_) / 128, 256, 0, stream>>>(h0_out, H_, g1Wih, H_, g1bih, gi, 3 * H_, H_);
  gemm_bf16_kernel<<<(3 * H_) / 128, 256, 0, stream>>>(hs1, H_, g1Whh, H_, g1bhh, gh, 3 * H_, H_);
  gru_gates_kernel<<<(B_ * H_ + 255) / 256, 256, 0, stream>>>(gi, gh, hs1, h1_out);

  // logits = [h1|ctx|emb] @ Wout^T + bout
  concat_out_kernel<<<(B_ * (H_ + E_ + EMB_) + 255) / 256, 256, 0, stream>>>(h1_out, ctx, emb, xo);
  gemm_bf16_kernel<<<V_ / 128, 256, 0, stream>>>(xo, H_ + E_ + EMB_, Wout, H_ + E_ + EMB_, bout, logits, V_, H_ + E_ + EMB_);
}
